// VAE_11613591568667
// MI455X (gfx1250) — compile-verified, matrix operand order checked by reading, not run on hardware
//
#include <hip/hip_runtime.h>

#define Nn   8192
#define Ee   262144
#define DIN  256
#define DH   256
#define DOUT 64

typedef __attribute__((ext_vector_type(16))) __bf16 v16bf;
typedef __attribute__((ext_vector_type(8)))  float  v8f;

union FragAB { v16bf v; unsigned u[8]; };

__device__ __forceinline__ unsigned short f2bf(float f) {
  unsigned u = __float_as_uint(f);
  u += 0x7FFFu + ((u >> 16) & 1u);          // round-to-nearest-even
  return (unsigned short)(u >> 16);
}

// ================= CSR construction (dst-major) =================
__global__ void k_zero_cnt(int* __restrict__ cnt) {
  int i = blockIdx.x * blockDim.x + threadIdx.x;
  if (i < Nn) cnt[i] = 0;
}

__global__ void k_count(const int* __restrict__ ei, int* __restrict__ cnt) {
  int e = blockIdx.x * blockDim.x + threadIdx.x;
  if (e < Ee) atomicAdd(&cnt[ei[Ee + e]], 1);
}

// deg = in-degree + self-loop
__global__ void k_dinv(const int* __restrict__ cnt, float* __restrict__ dinv) {
  int i = blockIdx.x * blockDim.x + threadIdx.x;
  if (i < Nn) dinv[i] = rsqrtf((float)cnt[i] + 1.0f);
}

// single-workgroup exclusive scan of 8192 counts -> rowstart[Nn+1], cursor copy
__global__ void k_scan(const int* __restrict__ cnt, int* __restrict__ rowstart,
                       int* __restrict__ cursor) {
  __shared__ int lds[1024];
  int t = threadIdx.x;
  int base = t * 8;
  int loc[8];
  int s = 0;
#pragma unroll
  for (int j = 0; j < 8; ++j) { loc[j] = s; s += cnt[base + j]; }
  lds[t] = s;
  __syncthreads();
  for (int ofs = 1; ofs < 1024; ofs <<= 1) {
    int v = (t >= ofs) ? lds[t - ofs] : 0;
    __syncthreads();
    lds[t] += v;
    __syncthreads();
  }
  int excl = (t == 0) ? 0 : lds[t - 1];
#pragma unroll
  for (int j = 0; j < 8; ++j) {
    int v = excl + loc[j];
    rowstart[base + j] = v;
    cursor[base + j]   = v;
  }
  if (t == 1023) rowstart[Nn] = lds[1023];
}

__global__ void k_fill_csr(const int* __restrict__ ei, int* __restrict__ cursor,
                           int* __restrict__ csr_src) {
  int e = blockIdx.x * blockDim.x + threadIdx.x;
  if (e < Ee) {
    int pos = atomicAdd(&cursor[ei[Ee + e]], 1);
    csr_src[pos] = ei[e];
  }
}

// ================= precision conversion =================
__global__ void k_f32_to_bf16(const float* __restrict__ src,
                              unsigned short* __restrict__ dst, int n) {
  int i = blockIdx.x * blockDim.x + threadIdx.x;
  if (i < n) dst[i] = f2bf(src[i]);
}

// W [K, Nc] f32 row-major -> Wt [Nc, K] bf16 (B-fragments load contiguous K pairs)
__global__ void k_w_transpose_bf16(const float* __restrict__ W,
                                   unsigned short* __restrict__ Wt, int K, int Nc) {
  int i = blockIdx.x * blockDim.x + threadIdx.x;
  if (i < K * Nc) {
    int k = i / Nc, n = i % Nc;
    Wt[n * K + k] = f2bf(W[i]);
  }
}

// ================= WMMA GEMM: one 16x16 tile per wave =================
__global__ void k_gemm_bf16_wmma(const unsigned short* __restrict__ A,
                                 const unsigned short* __restrict__ Bt,
                                 const float* __restrict__ bias,
                                 float* __restrict__ outF,
                                 unsigned short* __restrict__ outB,
                                 int M, int Nout, int K, int doRelu) {
  int wave = blockIdx.x * (blockDim.x >> 5) + (threadIdx.x >> 5);
  int lane = threadIdx.x & 31;
  int tilesN = Nout >> 4;
  int tn = wave % tilesN;
  int tm = wave / tilesN;
  if (tm >= (M >> 4)) return;               // wave-uniform
  int h = lane >> 4, l16 = lane & 15;
  int arow = (tm << 4) + l16;
  int bcol = (tn << 4) + l16;
  v8f acc = {};
  for (int k0 = 0; k0 < K; k0 += 32) {
    FragAB a, b;
#pragma unroll
    for (int v = 0; v < 8; ++v) {
      int g = v >> 2, vv = v & 3;
      int k = k0 + (g << 4) + (h << 3) + (vv << 1);
      a.u[v] = *reinterpret_cast<const unsigned*>(A  + (size_t)arow * K + k);
      b.u[v] = *reinterpret_cast<const unsigned*>(Bt + (size_t)bcol * K + k);
    }
    acc = __builtin_amdgcn_wmma_f32_16x16x32_bf16(false, a.v, false, b.v,
                                                  (short)0, acc, false, false);
  }
#pragma unroll
  for (int r = 0; r < 8; ++r) {
    int row = (tm << 4) + r + (h << 3);
    int col = (tn << 4) + l16;
    float vo = acc[r];
    if (bias)  vo += bias[col];
    if (doRelu) vo = fmaxf(vo, 0.0f);
    size_t off = (size_t)row * Nout + col;
    if (outF) outF[off] = vo;
    if (outB) outB[off] = f2bf(vo);
  }
}

// ================= fused GCN aggregation (gather, no atomics) =================
// One 256-thread block per node row (D == 256): self-loop + edge gather + bias
// + ReLU + bf16 store, all in registers.
__global__ void k_gcn_gather(const float* __restrict__ xw,
                             const int* __restrict__ rowstart,
                             const int* __restrict__ csr_src,
                             const float* __restrict__ dinv,
                             const float* __restrict__ bias,
                             unsigned short* __restrict__ outb) {
  int row = blockIdx.x;
  int t = threadIdx.x;                      // 0..255
  float dv = dinv[row];
  float acc = xw[(size_t)row * DH + t] * dv * dv;     // self-loop
  int beg = rowstart[row], end = rowstart[row + 1];
  for (int j = beg; j < end; ++j) {
    int s = csr_src[j];
    acc += xw[(size_t)s * DH + t] * (dinv[s] * dv);
  }
  outb[(size_t)row * DH + t] = f2bf(fmaxf(acc + bias[t], 0.0f));
}

// Heads fused: aggregate mean and logstd together, then z = noise*exp(ls)+m -> bf16.
__global__ void k_head_gather_z(const float* __restrict__ xm, const float* __restrict__ xl,
                                const int* __restrict__ rowstart,
                                const int* __restrict__ csr_src,
                                const float* __restrict__ dinv,
                                const float* __restrict__ bM, const float* __restrict__ bL,
                                const float* __restrict__ noise,
                                unsigned short* __restrict__ zb) {
  int row = blockIdx.x;
  int t = threadIdx.x;                      // 0..63
  float dv = dinv[row];
  float am = xm[(size_t)row * DOUT + t] * dv * dv;
  float al = xl[(size_t)row * DOUT + t] * dv * dv;
  int beg = rowstart[row], end = rowstart[row + 1];
  for (int j = beg; j < end; ++j) {
    int s = csr_src[j];
    float nrm = dinv[s] * dv;
    am += xm[(size_t)s * DOUT + t] * nrm;
    al += xl[(size_t)s * DOUT + t] * nrm;
  }
  float m  = fmaxf(am + bM[t], 0.0f);
  float ls = fmaxf(al + bL[t], 0.0f);
  size_t off = (size_t)row * DOUT + t;
  zb[off] = f2bf(noise[off] * __expf(ls) + m);
}

// ================= decoder: triu(sigmoid(z z^T), 1), NT streamed =================
__global__ void k_decode_wmma(const unsigned short* __restrict__ Zb,
                              float* __restrict__ out) {
  int wave = blockIdx.x * (blockDim.x >> 5) + (threadIdx.x >> 5);
  int lane = threadIdx.x & 31;
  const int tilesN = Nn >> 4;               // 512
  int tn = wave % tilesN;
  int tm = wave / tilesN;
  int h = lane >> 4, l16 = lane & 15;
  if (tn < tm) {                            // tile entirely at/below diagonal -> zeros
#pragma unroll
    for (int r = 0; r < 8; ++r) {
      size_t off = (size_t)((tm << 4) + r + (h << 3)) * Nn + (tn << 4) + l16;
      __builtin_nontemporal_store(0.0f, out + off);
    }
    return;
  }
  v8f acc = {};
  int arow = (tm << 4) + l16;
  int bcol = (tn << 4) + l16;
#pragma unroll
  for (int k0 = 0; k0 < DOUT; k0 += 32) {
    FragAB a, b;
#pragma unroll
    for (int v = 0; v < 8; ++v) {
      int g = v >> 2, vv = v & 3;
      int k = k0 + (g << 4) + (h << 3) + (vv << 1);
      a.u[v] = *reinterpret_cast<const unsigned*>(Zb + (size_t)arow * DOUT + k);
      b.u[v] = *reinterpret_cast<const unsigned*>(Zb + (size_t)bcol * DOUT + k);
    }
    acc = __builtin_amdgcn_wmma_f32_16x16x32_bf16(false, a.v, false, b.v,
                                                  (short)0, acc, false, false);
  }
#pragma unroll
  for (int r = 0; r < 8; ++r) {
    int row = (tm << 4) + r + (h << 3);
    int col = (tn << 4) + l16;
    float s = (col > row) ? 1.0f / (1.0f + __expf(-acc[r])) : 0.0f;
    __builtin_nontemporal_store(s, out + (size_t)row * Nn + col);
  }
}

extern "C" void kernel_launch(void* const* d_in, const int* in_sizes, int n_in,
                              void* d_out, int out_size, void* d_ws, size_t ws_size,
                              hipStream_t stream) {
  (void)in_sizes; (void)n_in; (void)out_size; (void)ws_size;
  const float* x        = (const float*)d_in[0];
  const int*   ei       = (const int*)  d_in[1];
  const float* noise    = (const float*)d_in[2];
  const float* mlp_w    = (const float*)d_in[3];
  const float* mlp_b    = (const float*)d_in[4];
  const float* w1       = (const float*)d_in[5];
  const float* b1       = (const float*)d_in[6];
  const float* w2       = (const float*)d_in[7];
  const float* b2       = (const float*)d_in[8];
  const float* w_mean   = (const float*)d_in[9];
  const float* b_mean   = (const float*)d_in[10];
  const float* w_logstd = (const float*)d_in[11];
  const float* b_logstd = (const float*)d_in[12];
  float* out = (float*)d_out;

  char* ws = (char*)d_ws;
  auto alloc = [&](size_t bytes) -> char* {
    char* p = ws; ws += (bytes + 255) & ~(size_t)255; return p;
  };
  int*            cnt      = (int*)alloc((size_t)Nn * 4);
  int*            rowstart = (int*)alloc((size_t)(Nn + 1) * 4);
  int*            cursor   = (int*)alloc((size_t)Nn * 4);
  int*            csr_src  = (int*)alloc((size_t)Ee * 4);
  float*          dinv     = (float*)alloc((size_t)Nn * 4);
  unsigned short* xb   = (unsigned short*)alloc((size_t)Nn * DIN * 2);
  unsigned short* wt0  = (unsigned short*)alloc((size_t)DH * DIN * 2);
  unsigned short* wt1  = (unsigned short*)alloc((size_t)DH * DH * 2);
  unsigned short* wt2  = (unsigned short*)alloc((size_t)DH * DH * 2);
  unsigned short* wtm  = (unsigned short*)alloc((size_t)DOUT * DH * 2);
  unsigned short* wtl  = (unsigned short*)alloc((size_t)DOUT * DH * 2);
  unsigned short* h0b  = (unsigned short*)alloc((size_t)Nn * DH * 2);
  unsigned short* h1b  = (unsigned short*)alloc((size_t)Nn * DH * 2);
  unsigned short* h2b  = (unsigned short*)alloc((size_t)Nn * DH * 2);
  float*          tmp  = (float*)alloc((size_t)Nn * DH * 4);
  float*          tmpM = (float*)alloc((size_t)Nn * DOUT * 4);
  float*          tmpL = (float*)alloc((size_t)Nn * DOUT * 4);
  unsigned short* zb   = (unsigned short*)alloc((size_t)Nn * DOUT * 2);

  dim3 B(256);
  // ---- CSR build + normalization (int atomics only) ----
  k_zero_cnt<<<Nn / 256, B, 0, stream>>>(cnt);
  k_count   <<<Ee / 256, B, 0, stream>>>(ei, cnt);
  k_dinv    <<<Nn / 256, B, 0, stream>>>(cnt, dinv);
  k_scan    <<<1, 1024, 0, stream>>>(cnt, rowstart, cursor);
  k_fill_csr<<<Ee / 256, B, 0, stream>>>(ei, cursor, csr_src);

  // ---- conversions ----
  k_f32_to_bf16<<<(Nn * DIN) / 256, B, 0, stream>>>(x, xb, Nn * DIN);
  k_w_transpose_bf16<<<(DIN * DH)  / 256, B, 0, stream>>>(mlp_w,    wt0, DIN, DH);
  k_w_transpose_bf16<<<(DH  * DH)  / 256, B, 0, stream>>>(w1,       wt1, DH,  DH);
  k_w_transpose_bf16<<<(DH  * DH)  / 256, B, 0, stream>>>(w2,       wt2, DH,  DH);
  k_w_transpose_bf16<<<(DH  * DOUT)/ 256, B, 0, stream>>>(w_mean,   wtm, DH,  DOUT);
  k_w_transpose_bf16<<<(DH  * DOUT)/ 256, B, 0, stream>>>(w_logstd, wtl, DH,  DOUT);

  const int gWide = (Nn / 16) * (DH / 16) / 8;    // 1024 blocks
  const int gHead = (Nn / 16) * (DOUT / 16) / 8;  // 256 blocks

  // layer 0: h0 = x @ mlp_w + mlp_b (bf16)
  k_gemm_bf16_wmma<<<gWide, B, 0, stream>>>(xb, wt0, mlp_b, nullptr, h0b, Nn, DH, DIN, 0);

  // GCN layer 1: xw -> gather-aggregate (fused bias+relu+bf16)
  k_gemm_bf16_wmma<<<gWide, B, 0, stream>>>(h0b, wt1, nullptr, tmp, nullptr, Nn, DH, DH, 0);
  k_gcn_gather<<<Nn, B, 0, stream>>>(tmp, rowstart, csr_src, dinv, b1, h1b);

  // GCN layer 2
  k_gemm_bf16_wmma<<<gWide, B, 0, stream>>>(h1b, wt2, nullptr, tmp, nullptr, Nn, DH, DH, 0);
  k_gcn_gather<<<Nn, B, 0, stream>>>(tmp, rowstart, csr_src, dinv, b2, h2b);

  // heads: two GEMMs, one fused gather producing z (bf16)
  k_gemm_bf16_wmma<<<gHead, B, 0, stream>>>(h2b, wtm, nullptr, tmpM, nullptr, Nn, DOUT, DH, 0);
  k_gemm_bf16_wmma<<<gHead, B, 0, stream>>>(h2b, wtl, nullptr, tmpL, nullptr, Nn, DOUT, DH, 0);
  k_head_gather_z<<<Nn, 64, 0, stream>>>(tmpM, tmpL, rowstart, csr_src, dinv,
                                         b_mean, b_logstd, noise, zb);

  // decoder: 512x512 tiles, 8 waves/block
  k_decode_wmma<<<(Nn / 16) * (Nn / 16) / 8, B, 0, stream>>>(zb, out);
}
